// TisaTransformer_17557826306458
// MI455X (gfx1250) — matrix operand order repaired and checked by reading, not compiled, not run on hardware
//
#include <hip/hip_runtime.h>

// Problem constants (match reference)
#define BB   2
#define SS   2048
#define DIN  256
#define DD   1024
#define HH   16
#define DH_  64
#define FF_  4096
#define KK   21
#define NBLK 2
#define DOUT 256
#define MTOK (BB * SS)   // 4096 token rows

typedef __attribute__((ext_vector_type(16))) __bf16 v16bf;
typedef __attribute__((ext_vector_type(8)))  float  v8f;
typedef __attribute__((ext_vector_type(4)))  unsigned int u32x4;
typedef __attribute__((ext_vector_type(8)))  unsigned int u32x8;

union AFrag { v16bf v; unsigned int u[8]; };

// A-fragment k index for 16-bit WMMA (16x32): VGPR v, lane-half hf
__device__ __forceinline__ int kA(int v, int hf) {
    return (v < 4 ? 2 * v : 8 + 2 * v) + hf * 8;
}

__device__ __forceinline__ float hmax16(float x) {
    #pragma unroll
    for (int m = 1; m < 16; m <<= 1) x = fmaxf(x, __shfl_xor(x, m, 32));
    return x;
}
__device__ __forceinline__ float hsum16(float x) {
    #pragma unroll
    for (int m = 1; m < 16; m <<= 1) x += __shfl_xor(x, m, 32);
    return x;
}

// Generic pointer to LDS byte offset (low 32 bits of shared-aperture address)
__device__ __forceinline__ unsigned lds_off(const void* p) {
    return (unsigned)(size_t)p;
}

// ---------------------------------------------------------------------------
// Tensor Data Mover: async 2D tile load (32 x 128, bf16) global -> LDS with
// row padding (64B data + 16B pad -> LDS row stride 80B == [128][40] bf16).
// D# packing per CDNA5 ISA. Issued per-wave
// (EXEC ignored), tracked by TENSORcnt.
// ---------------------------------------------------------------------------
__device__ __forceinline__ void tdm_tile_32x128(
    unsigned lds_byte_off, const __bf16* gsrc,
    unsigned tensor_d0, unsigned tensor_d1, unsigned long long stride_el)
{
    unsigned long long ga = (unsigned long long)(size_t)gsrc;
    u32x4 g0;
    g0[0] = 1u;                                   // count=1 (valid descriptor)
    g0[1] = lds_byte_off;                         // lds_addr
    g0[2] = (unsigned)ga;                         // global_addr[31:0]
    g0[3] = (unsigned)(ga >> 32) | 0x80000000u;   // global_addr[56:32] | type=2
    u32x8 g1;
    // data_size=2B, pad_enable, pad_interval=16 DWORDs, pad_amount=4 DWORDs
    g1[0] = (1u << 16) | (1u << 20) | (3u << 22) | (3u << 25);
    g1[1] = (tensor_d0 & 0xFFFFu) << 16;                          // tensor_dim0 lo
    g1[2] = (tensor_d0 >> 16) | ((tensor_d1 & 0xFFFFu) << 16);    // dim0 hi | dim1 lo
    g1[3] = (tensor_d1 >> 16) | (32u << 16);                      // dim1 hi | tile_dim0=32
    g1[4] = 128u;                                                 // tile_dim1=128, tile_dim2=0
    g1[5] = (unsigned)stride_el;                                  // tensor_dim0_stride lo
    g1[6] = (unsigned)(stride_el >> 32) & 0xFFFFu;                // stride hi | dim1_stride=0
    g1[7] = 0u;
    asm volatile("tensor_load_to_lds %0, %1" :: "s"(g0), "s"(g1) : "memory");
}

// ---------------------------------------------------------------------------
// f32 -> bf16 conversion (row-major copy)
// ---------------------------------------------------------------------------
__global__ void f32_to_bf16(const float* __restrict__ in, __bf16* __restrict__ out, int n) {
    int i = blockIdx.x * 256 + threadIdx.x;
    if (i < n) out[i] = (__bf16)in[i];
}

// f32 (K,N) -> bf16 transposed (N,K) : one-time weight preconditioning so all
// GEMM B-operands become [N][K] tiles loadable by TDM with no in-loop shuffle.
__global__ __launch_bounds__(256) void f32_to_bf16_T(
    const float* __restrict__ in, __bf16* __restrict__ out, int K, int N)
{
    __shared__ float tile[32][33];
    const int n0 = blockIdx.x * 32, k0 = blockIdx.y * 32;
    const int tx = threadIdx.x & 31, ty = threadIdx.x >> 5;   // 8 rows of 32
    #pragma unroll
    for (int j = 0; j < 32; j += 8)
        tile[ty + j][tx] = in[(size_t)(k0 + ty + j) * N + n0 + tx];
    __syncthreads();
    #pragma unroll
    for (int j = 0; j < 32; j += 8)
        out[(size_t)(n0 + ty + j) * K + k0 + tx] = (__bf16)tile[tx][ty + j];
}

// ---------------------------------------------------------------------------
// Tiled bf16 WMMA GEMM: C[M,N] = A[M,K] * Bt[N,K]^T (+bias), fused epilogue.
// EPI: 0 = +bias, f32 out; 1 = bf16 out; 2 = GELU(exact), bf16 out; 3 = f32 out
// Block 256 threads (8 waves), tile 128x128, K-step 32.
// TDM double-buffered LDS staging: wave 0 issues next tile while all waves
// run WMMA on the current one.
// ---------------------------------------------------------------------------
template <int EPI>
__global__ __launch_bounds__(256) void gemm_bf16(
    const __bf16* __restrict__ A,    // [M][Kd] row-major (bf16 activations)
    const __bf16* __restrict__ Bt,   // [N][Kd] row-major (pre-transposed weights)
    const float* __restrict__ bias, void* __restrict__ Cout,
    int M, int N, int Kd)
{
    __shared__ __bf16 As[2][128][40];   // [buf][m][k], row stride 80B (TDM-padded)
    __shared__ __bf16 Bs[2][128][40];   // [buf][n][k]

    const int t    = threadIdx.x;
    const int lane = t & 31, wave = t >> 5;
    const int hf   = lane >> 4, l16 = lane & 15;
    const int wm   = (wave >> 1) * 32;   // wave M offset within 128
    const int wn   = (wave & 1) * 64;    // wave N offset within 128
    const int m0   = blockIdx.y * 128, n0 = blockIdx.x * 128;
    const int nIter = Kd >> 5;

    v8f acc[2][4];
    #pragma unroll
    for (int mt = 0; mt < 2; ++mt)
        #pragma unroll
        for (int nt = 0; nt < 4; ++nt)
            #pragma unroll
            for (int v = 0; v < 8; ++v) acc[mt][nt][v] = 0.0f;

    // Prologue: stage tile 0 into buffer 0 via TDM (wave 0 only)
    if (wave == 0) {
        tdm_tile_32x128(lds_off(&As[0][0][0]), A + (size_t)m0 * Kd,
                        (unsigned)Kd, 128u, (unsigned long long)Kd);
        tdm_tile_32x128(lds_off(&Bs[0][0][0]), Bt + (size_t)n0 * Kd,
                        (unsigned)Kd, 128u, (unsigned long long)Kd);
    }

    for (int it = 0; it < nIter; ++it) {
        const int buf = it & 1;
        if (wave == 0) {
            if (it + 1 < nIter) {
                const int k0 = (it + 1) * 32;
                tdm_tile_32x128(lds_off(&As[buf ^ 1][0][0]),
                                A + (size_t)m0 * Kd + k0,
                                (unsigned)(Kd - k0), 128u, (unsigned long long)Kd);
                tdm_tile_32x128(lds_off(&Bs[buf ^ 1][0][0]),
                                Bt + (size_t)n0 * Kd + k0,
                                (unsigned)(Kd - k0), 128u, (unsigned long long)Kd);
                __builtin_amdgcn_s_wait_tensorcnt(2);  // current tile done
            } else {
                __builtin_amdgcn_s_wait_tensorcnt(0);
            }
        }
        __syncthreads();   // tile `it` visible to all waves

        AFrag af[2], bfr[4];
        #pragma unroll
        for (int mt = 0; mt < 2; ++mt) {
            int mr = wm + mt * 16 + l16;
            #pragma unroll
            for (int v = 0; v < 8; ++v)
                af[mt].u[v] = *(const unsigned int*)&As[buf][mr][kA(v, hf)];
        }
        #pragma unroll
        for (int nt = 0; nt < 4; ++nt) {
            int nr = wn + nt * 16 + l16;
            #pragma unroll
            for (int v = 0; v < 8; ++v)
                bfr[nt].u[v] = *(const unsigned int*)&Bs[buf][nr][kA(v, hf)];
        }
        #pragma unroll
        for (int mt = 0; mt < 2; ++mt)
            #pragma unroll
            for (int nt = 0; nt < 4; ++nt)
                acc[mt][nt] = __builtin_amdgcn_wmma_f32_16x16x32_bf16(
                    false, af[mt].v, false, bfr[nt].v, (short)0, acc[mt][nt], false, false);
        __syncthreads();   // everyone done reading buf before TDM overwrites it
    }

    // ---- epilogue
    #pragma unroll
    for (int mt = 0; mt < 2; ++mt)
        #pragma unroll
        for (int nt = 0; nt < 4; ++nt)
            #pragma unroll
            for (int v = 0; v < 8; ++v) {
                int row = m0 + wm + mt * 16 + v + 8 * hf;
                int col = n0 + wn + nt * 16 + l16;
                float x = acc[mt][nt][v];
                if (EPI == 0) x += bias[col];
                if (EPI == 2) x = 0.5f * x * (1.0f + erff(x * 0.70710678118f));
                size_t off = (size_t)row * N + col;
                if (EPI == 1 || EPI == 2) ((__bf16*)Cout)[off] = (__bf16)x;
                else                      ((float*)Cout)[off]  = x;
            }
}

// ---------------------------------------------------------------------------
// TISA bias table: table[h][d], d in [0, 2S-2], sum of K Gaussians
// ---------------------------------------------------------------------------
__global__ void tisa_table(const float* __restrict__ amp, const float* __restrict__ sharp,
                           const float* __restrict__ off, float* __restrict__ table)
{
    int d = blockIdx.x * 256 + threadIdx.x;
    int h = blockIdx.y;
    if (d >= 2 * SS - 1) return;
    float dv = (float)(d - (SS - 1));
    float s = 0.0f;
    #pragma unroll
    for (int k = 0; k < KK; ++k) {
        float a  = amp[h * KK + k];
        float sh = fabsf(sharp[h * KK + k]);
        float of = off[h * KK + k];
        float dd = of - dv;
        s += a * __expf(-sh * dd * dd);
    }
    table[(size_t)h * (2 * SS - 1) + d] = s;
}

// ---------------------------------------------------------------------------
// Flash attention with TISA bias. qkv rows = tokens (B*S), cols = 3D with
// layout [K | V | Q]; head h occupies cols h*64..h*64+63 of each segment.
// Grid (S/128, H, B), 256 threads = 8 waves; each wave owns a 16-row Q strip.
// Output o is bf16 (B*S, D).
// ---------------------------------------------------------------------------
__global__ __launch_bounds__(256) void flash_attn(
    const __bf16* __restrict__ qkv, const float* __restrict__ table,
    __bf16* __restrict__ o)
{
    __shared__ __bf16 Ksh[64][72];       // [key][dh]
    __shared__ __bf16 Vt[64][72];        // [dh][key] (transposed)
    __shared__ __bf16 Psh[8][16][72];    // per-wave P staging [row][key]

    const int t    = threadIdx.x;
    const int lane = t & 31, wave = t >> 5;
    const int hf   = lane >> 4, l16 = lane & 15;
    const int b    = blockIdx.z, h = blockIdx.y;
    const int s0   = blockIdx.x * 128 + wave * 16;    // this wave's first q row
    const size_t rs = 3 * DD;
    const float* tbl = table + (size_t)h * (2 * SS - 1);

    // Q strip as A-fragments (2 K-chunks of 32 over DH=64), kept in registers
    AFrag qf[2];
    {
        const __bf16* qrow = qkv + (size_t)(b * SS + s0 + l16) * rs + 2 * DD + h * DH_;
        #pragma unroll
        for (int kc = 0; kc < 2; ++kc)
            #pragma unroll
            for (int v = 0; v < 8; ++v)
                qf[kc].u[v] = *(const unsigned int*)(qrow + 32 * kc + kA(v, hf));
    }

    v8f oacc[4];
    #pragma unroll
    for (int nt = 0; nt < 4; ++nt)
        #pragma unroll
        for (int v = 0; v < 8; ++v) oacc[nt][v] = 0.0f;
    float rowm[8], rowsum[8];
    #pragma unroll
    for (int v = 0; v < 8; ++v) { rowm[v] = -3.0e38f; rowsum[v] = 0.0f; }

    for (int kb = 0; kb < SS; kb += 64) {
        // ---- stage K (64x64) and V (transposed) tiles
        #pragma unroll
        for (int i = 0; i < 2; ++i) {
            int idx = t * 2 + i;
            int key = idx >> 3, dc = (idx & 7) * 8;
            size_t base = (size_t)(b * SS + kb + key) * rs + h * DH_ + dc;
            *(uint4*)&Ksh[key][dc] = *(const uint4*)(qkv + base);           // K segment
            union { uint4 q; __bf16 e[8]; } u;
            u.q = *(const uint4*)(qkv + base + DD);                         // V segment
            #pragma unroll
            for (int j = 0; j < 8; ++j) Vt[dc + j][key] = u.e[j];
        }
        if (kb + 64 < SS)   // global_prefetch_b8 next K tile
            __builtin_prefetch(qkv + (size_t)(b * SS + kb + 64 + (t >> 2)) * rs + h * DH_, 0, 1);
        __syncthreads();

        // ---- scores S = Q K^T / sqrt(DH) + bias    (16x64 strip per wave)
        float s[4][8];
        #pragma unroll
        for (int nt = 0; nt < 4; ++nt) {
            v8f sc;
            #pragma unroll
            for (int v = 0; v < 8; ++v) sc[v] = 0.0f;
            #pragma unroll
            for (int kc = 0; kc < 2; ++kc) {
                AFrag bf2;
                int key = nt * 16 + l16;
                #pragma unroll
                for (int v = 0; v < 8; ++v)
                    bf2.u[v] = *(const unsigned int*)&Ksh[key][kA(v, hf) + 32 * kc];
                sc = __builtin_amdgcn_wmma_f32_16x16x32_bf16(
                    false, qf[kc].v, false, bf2.v, (short)0, sc, false, false);
            }
            #pragma unroll
            for (int v = 0; v < 8; ++v) {
                int kg  = kb + nt * 16 + l16;
                int rel = kg - (s0 + v + 8 * hf) + (SS - 1);
                s[nt][v] = sc[v] * 0.125f + tbl[rel];
            }
        }

        // ---- online softmax (per row v + 8*hf; 16-lane reductions)
        float pscale[8];
        #pragma unroll
        for (int v = 0; v < 8; ++v) {
            float mx = fmaxf(fmaxf(s[0][v], s[1][v]), fmaxf(s[2][v], s[3][v]));
            mx = hmax16(mx);
            float newm  = fmaxf(rowm[v], mx);
            float scale = __expf(rowm[v] - newm);
            float ps = 0.0f;
            #pragma unroll
            for (int nt = 0; nt < 4; ++nt) {
                float p = __expf(s[nt][v] - newm);
                s[nt][v] = p; ps += p;
            }
            ps = hsum16(ps);
            rowsum[v] = rowsum[v] * scale + ps;
            rowm[v]   = newm;
            pscale[v] = scale;
        }
        #pragma unroll
        for (int nt = 0; nt < 4; ++nt)
            #pragma unroll
            for (int v = 0; v < 8; ++v) oacc[nt][v] *= pscale[v];

        // ---- stage P (C-layout -> LDS), reload as A-fragments (wave-local,
        //      same-wave DS ops are in-order)
        #pragma unroll
        for (int nt = 0; nt < 4; ++nt)
            #pragma unroll
            for (int v = 0; v < 8; ++v)
                Psh[wave][v + 8 * hf][nt * 16 + l16] = (__bf16)s[nt][v];
        AFrag pf[2];
        #pragma unroll
        for (int kc = 0; kc < 2; ++kc)
            #pragma unroll
            for (int v = 0; v < 8; ++v)
                pf[kc].u[v] = *(const unsigned int*)&Psh[wave][l16][kA(v, hf) + 32 * kc];

        // ---- O += P V
        #pragma unroll
        for (int nt = 0; nt < 4; ++nt) {
            #pragma unroll
            for (int kc = 0; kc < 2; ++kc) {
                AFrag vf;
                int dh = nt * 16 + l16;
                #pragma unroll
                for (int v = 0; v < 8; ++v)
                    vf.u[v] = *(const unsigned int*)&Vt[dh][kA(v, hf) + 32 * kc];
                oacc[nt] = __builtin_amdgcn_wmma_f32_16x16x32_bf16(
                    false, pf[kc].v, false, vf.v, (short)0, oacc[nt], false, false);
            }
        }
        __syncthreads();
    }

    // ---- normalize and write O (bf16), layout (B*S, D) with head h block
    #pragma unroll
    for (int nt = 0; nt < 4; ++nt)
        #pragma unroll
        for (int v = 0; v < 8; ++v) {
            int row = b * SS + s0 + v + 8 * hf;
            int col = h * DH_ + nt * 16 + l16;
            o[(size_t)row * DD + col] = (__bf16)(oacc[nt][v] / rowsum[v]);
        }
}

// ---------------------------------------------------------------------------
// Row-wise reductions for the residual paths
// ---------------------------------------------------------------------------
__device__ __forceinline__ float block_sum(float v, float* red) {
    int t = threadIdx.x;
    red[t] = v; __syncthreads();
    for (int s = 128; s > 0; s >>= 1) { if (t < s) red[t] += red[t + s]; __syncthreads(); }
    float r = red[0]; __syncthreads();
    return r;
}

// h = LN(a * sigmoid(bg) + h) ; g row = [a(0:1024) | bg(1024:2048)] (bias fused)
__global__ __launch_bounds__(256) void gate_ln(
    const float* __restrict__ g, const float* __restrict__ w,
    const float* __restrict__ bb, float* __restrict__ h)
{
    __shared__ float red[256];
    const int row = blockIdx.x, t = threadIdx.x;
    const float* gr = g + (size_t)row * (2 * DD);
    float* hr = h + (size_t)row * DD;
    float x[4]; float s = 0.0f;
    #pragma unroll
    for (int j = 0; j < 4; ++j) {
        int c = t + j * 256;
        float a = gr[c], bg = gr[DD + c];
        float v = a / (1.0f + __expf(-bg)) + hr[c];
        x[j] = v; s += v;
    }
    float mean = block_sum(s, red) * (1.0f / DD);
    float s2 = 0.0f;
    #pragma unroll
    for (int j = 0; j < 4; ++j) { float d = x[j] - mean; s2 += d * d; }
    float var = block_sum(s2, red) * (1.0f / DD);
    float r = rsqrtf(var + 1e-5f);
    #pragma unroll
    for (int j = 0; j < 4; ++j) {
        int c = t + j * 256;
        hr[c] = (x[j] - mean) * r * w[c] + bb[c];
    }
}

// h = LN(f + h)
__global__ __launch_bounds__(256) void add_ln(
    const float* __restrict__ f, const float* __restrict__ w,
    const float* __restrict__ bb, float* __restrict__ h)
{
    __shared__ float red[256];
    const int row = blockIdx.x, t = threadIdx.x;
    const float* fr = f + (size_t)row * DD;
    float* hr = h + (size_t)row * DD;
    float x[4]; float s = 0.0f;
    #pragma unroll
    for (int j = 0; j < 4; ++j) {
        int c = t + j * 256;
        float v = fr[c] + hr[c];
        x[j] = v; s += v;
    }
    float mean = block_sum(s, red) * (1.0f / DD);
    float s2 = 0.0f;
    #pragma unroll
    for (int j = 0; j < 4; ++j) { float d = x[j] - mean; s2 += d * d; }
    float var = block_sum(s2, red) * (1.0f / DD);
    float r = rsqrtf(var + 1e-5f);
    #pragma unroll
    for (int j = 0; j < 4; ++j) {
        int c = t + j * 256;
        hr[c] = (x[j] - mean) * r * w[c] + bb[c];
    }
}

// ---------------------------------------------------------------------------
// Host-side orchestration
// ---------------------------------------------------------------------------
static inline void cvt(const float* in, __bf16* out, int n, hipStream_t s) {
    f32_to_bf16<<<(n + 255) / 256, 256, 0, s>>>(in, out, n);
}
static inline void cvtT(const float* in, __bf16* out, int K, int N, hipStream_t s) {
    f32_to_bf16_T<<<dim3(N / 32, K / 32), 256, 0, s>>>(in, out, K, N);
}

extern "C" void kernel_launch(void* const* d_in, const int* in_sizes, int n_in,
                              void* d_out, int out_size, void* d_ws, size_t ws_size,
                              hipStream_t stream) {
    (void)in_sizes; (void)n_in; (void)out_size; (void)ws_size;
    const float* x      = (const float*)d_in[0];
    const float* W_in   = (const float*)d_in[1];
    const float* b_in   = (const float*)d_in[2];
    const float* W_out  = (const float*)d_in[3];
    const float* b_out  = (const float*)d_in[4];
    const float* W_qkv  = (const float*)d_in[5];
    const float* W_gate = (const float*)d_in[6];
    const float* b_gate = (const float*)d_in[7];
    const float* ln1_w  = (const float*)d_in[8];
    const float* ln1_b  = (const float*)d_in[9];
    const float* ln2_w  = (const float*)d_in[10];
    const float* ln2_b  = (const float*)d_in[11];
    const float* W_ff1  = (const float*)d_in[12];
    const float* W_ff2  = (const float*)d_in[13];
    const float* amp    = (const float*)d_in[14];
    const float* sharp  = (const float*)d_in[15];
    const float* off    = (const float*)d_in[16];
    float* out = (float*)d_out;

    char* ws = (char*)d_ws;
    size_t p = 0;
    auto alloc = [&](size_t bytes) -> void* {
        void* r = ws + p;
        p = (p + bytes + 255) & ~(size_t)255;
        return r;
    };
    __bf16* xb     = (__bf16*)alloc((size_t)MTOK * DIN * 2);
    __bf16* Wb_in  = (__bf16*)alloc((size_t)DIN * DD * 2);        // [DD][DIN]
    __bf16* Wb_qkv = (__bf16*)alloc((size_t)NBLK * DD * 3 * DD * 2); // [3DD][DD] per blk
    __bf16* Wb_gt  = (__bf16*)alloc((size_t)NBLK * DD * 2 * DD * 2); // [2DD][DD] per blk
    __bf16* Wb_f1  = (__bf16*)alloc((size_t)NBLK * DD * FF_ * 2);    // [FF][DD]  per blk
    __bf16* Wb_f2  = (__bf16*)alloc((size_t)NBLK * FF_ * DD * 2);    // [DD][FF]  per blk
    __bf16* Wb_out = (__bf16*)alloc((size_t)DD * DOUT * 2);          // [DOUT][DD]
    float*  hbuf   = (float*)alloc((size_t)MTOK * DD * 4);
    __bf16* hb     = (__bf16*)alloc((size_t)MTOK * DD * 2);
    __bf16* qkvb   = (__bf16*)alloc((size_t)MTOK * 3 * DD * 2);
    __bf16* ob     = (__bf16*)alloc((size_t)MTOK * DD * 2);
    float*  gbuf   = (float*)alloc((size_t)MTOK * 2 * DD * 4);
    __bf16* ffb    = (__bf16*)alloc((size_t)MTOK * FF_ * 2);
    float*  fbuf   = (float*)alloc((size_t)MTOK * DD * 4);
    float*  table  = (float*)alloc((size_t)HH * (2 * SS - 1) * 4);

    // One-time conversions; weights transposed to [N][K] for the TDM GEMM.
    cvt(x, xb, MTOK * DIN, stream);
    cvtT(W_in, Wb_in, DIN, DD, stream);
    for (int i = 0; i < NBLK; ++i) {
        cvtT(W_qkv  + (size_t)i * DD * 3 * DD, Wb_qkv + (size_t)i * DD * 3 * DD, DD, 3 * DD, stream);
        cvtT(W_gate + (size_t)i * DD * 2 * DD, Wb_gt  + (size_t)i * DD * 2 * DD, DD, 2 * DD, stream);
        cvtT(W_ff1  + (size_t)i * DD * FF_,    Wb_f1  + (size_t)i * DD * FF_,    DD, FF_,    stream);
        cvtT(W_ff2  + (size_t)i * FF_ * DD,    Wb_f2  + (size_t)i * FF_ * DD,    FF_, DD,    stream);
    }
    cvtT(W_out, Wb_out, DD, DOUT, stream);

    // h = x @ W_in + b_in
    gemm_bf16<0><<<dim3(DD / 128, MTOK / 128), 256, 0, stream>>>(
        xb, Wb_in, b_in, hbuf, MTOK, DD, DIN);

    for (int i = 0; i < NBLK; ++i) {
        cvt(hbuf, hb, MTOK * DD, stream);
        // qkv = h @ W_qkv[i]  (bf16 out for attention)
        gemm_bf16<1><<<dim3(3 * DD / 128, MTOK / 128), 256, 0, stream>>>(
            hb, Wb_qkv + (size_t)i * DD * 3 * DD, nullptr, qkvb, MTOK, 3 * DD, DD);
        // TISA bias table
        tisa_table<<<dim3((2 * SS - 1 + 255) / 256, HH), 256, 0, stream>>>(
            amp + i * HH * KK, sharp + i * HH * KK, off + i * HH * KK, table);
        // attention -> o (bf16)
        flash_attn<<<dim3(SS / 128, HH, BB), 256, 0, stream>>>(qkvb, table, ob);
        // g = o @ W_gate[i] + b_gate[i]
        gemm_bf16<0><<<dim3(2 * DD / 128, MTOK / 128), 256, 0, stream>>>(
            ob, Wb_gt + (size_t)i * DD * 2 * DD, b_gate + (size_t)i * 2 * DD,
            gbuf, MTOK, 2 * DD, DD);
        // h = LN2(a*sigmoid(bg) + h)
        gate_ln<<<MTOK, 256, 0, stream>>>(gbuf, ln2_w + i * DD, ln2_b + i * DD, hbuf);
        cvt(hbuf, hb, MTOK * DD, stream);
        // t = gelu(h @ W_ff1[i]) (bf16 out)
        gemm_bf16<2><<<dim3(FF_ / 128, MTOK / 128), 256, 0, stream>>>(
            hb, Wb_f1 + (size_t)i * DD * FF_, nullptr, ffb, MTOK, FF_, DD);
        // f = t @ W_ff2[i]
        gemm_bf16<3><<<dim3(DD / 128, MTOK / 128), 256, 0, stream>>>(
            ffb, Wb_f2 + (size_t)i * FF_ * DD, nullptr, fbuf, MTOK, DD, FF_);
        // h = LN1(f + h)
        add_ln<<<MTOK, 256, 0, stream>>>(fbuf, ln1_w + i * DD, ln1_b + i * DD, hbuf);
    }

    // out = h @ W_out + b_out
    cvt(hbuf, hb, MTOK * DD, stream);
    gemm_bf16<0><<<dim3(DOUT / 128, MTOK / 128), 256, 0, stream>>>(
        hb, Wb_out, b_out, out, MTOK, DOUT, DD);
}
